// MultiHeadAttention_824633721543
// MI455X (gfx1250) — compile-verified, matrix operand order checked by reading, not run on hardware
//
#include <hip/hip_runtime.h>
#include <hip/hip_bf16.h>

// ---------------------------------------------------------------------------
// MI455X (gfx1250) multi-head attention, wave32 + WMMA f32_16x16x32_f16,
// async global->LDS staging (ASYNCcnt) with double-buffered LDS tiles.
// D_MODEL=768, H=12, D_K=64, WINDOW=256, B=2, S=2048.
// Mask (faithful to reference): allowed iff (q%256) >= (k%256).
// ---------------------------------------------------------------------------

typedef __attribute__((ext_vector_type(4)))  _Float16 v4h;
typedef __attribute__((ext_vector_type(8)))  _Float16 v8h;
typedef __attribute__((ext_vector_type(16))) _Float16 v16h;
typedef __attribute__((ext_vector_type(8)))  float    v8f;

#define LDK 72            // padded LDS row stride (halves) for 64-wide K tiles
#define LDP 40            // padded LDS row stride (halves) for 32-wide tiles
#define NEGINF (-1e30f)

#define WMMA_F16(A, B, C) \
  __builtin_amdgcn_wmma_f32_16x16x32_f16(false, (A), false, (B), (short)0, (C), false, false)

// CDNA5 async copy: global memory -> LDS, tracked by ASYNCcnt (no VGPR data).
__device__ __forceinline__ void async_b128(const void* ldsPtr, const void* gptr) {
  unsigned lds = (unsigned)(unsigned long long)ldsPtr;     // LDS aperture low 32 bits
  unsigned long long ga = (unsigned long long)gptr;
  asm volatile("global_load_async_to_lds_b128 %0, %1, off"
               :: "v"(lds), "v"(ga) : "memory");
}
__device__ __forceinline__ void wait_async0() {
  asm volatile("s_wait_asynccnt 0x0" ::: "memory");
}

// Load one 16x32 (f16) A/B fragment from LDS per the documented CDNA5 layout:
// lane<16 holds K 0..7 & 16..23, lane>=16 holds K 8..15 & 24..31 (two 16B runs).
__device__ __forceinline__ v16h lds_frag(const _Float16* base, int row, int stride) {
  const int lane = threadIdx.x & 31;
  const _Float16* p = base + row * stride + ((lane < 16) ? 0 : 8);
  v8h lo = *(const v8h*)p;
  v8h hi = *(const v8h*)(p + 16);
  return __builtin_shufflevector(lo, hi, 0,1,2,3,4,5,6,7,8,9,10,11,12,13,14,15);
}

// Same layout sourced from global memory (resident Q fragments).
__device__ __forceinline__ v16h glb_frag(const _Float16* rowPtr) {
  const int lane = threadIdx.x & 31;
  const _Float16* p = rowPtr + ((lane < 16) ? 0 : 8);
  v8h lo = *(const v8h*)p;
  v8h hi = *(const v8h*)(p + 16);
  return __builtin_shufflevector(lo, hi, 0,1,2,3,4,5,6,7,8,9,10,11,12,13,14,15);
}

// ---------------------------------------------------------------------------
// fp32 -> f16 convert (vectorized x4)
// ---------------------------------------------------------------------------
__global__ __launch_bounds__(256) void cvt_f32_f16(const float* __restrict__ src,
                                                   _Float16* __restrict__ dst, int n) {
  int i = (blockIdx.x * 256 + threadIdx.x) * 4;
  if (i + 3 < n) {
    float4 f = *(const float4*)(src + i);
    v4h h = { (_Float16)f.x, (_Float16)f.y, (_Float16)f.z, (_Float16)f.w };
    *(v4h*)(dst + i) = h;
  }
}

// ---------------------------------------------------------------------------
// 128x128 GEMM core, K staged 64-wide via async-to-LDS, double buffered.
// 8 waves: wave computes 64x32 region = 4x2 WMMA accumulators.
// ---------------------------------------------------------------------------
__device__ __forceinline__ void stage_tiles(const _Float16* __restrict__ Ag,
                                            const _Float16* __restrict__ Bg,
                                            _Float16* sA, _Float16* sB, int k0) {
  const int t  = threadIdx.x;
  const int r0 = t >> 3;             // 0..31
  const int c8 = (t & 7) * 8;        // 0..56 halves
#pragma unroll
  for (int op = 0; op < 4; ++op) {
    const int r = op * 32 + r0;      // 0..127
    async_b128(&sA[r * LDK + c8], &Ag[(size_t)r * 768 + k0 + c8]);
    async_b128(&sB[r * LDK + c8], &Bg[(size_t)r * 768 + k0 + c8]);
  }
}

__device__ __forceinline__ void gemm128x128(const _Float16* __restrict__ Ag,
                                            const _Float16* __restrict__ Bg,
                                            _Float16* sA0, _Float16* sB0,
                                            _Float16* sA1, _Float16* sB1,
                                            v8f acc[4][2]) {
  const int lane = threadIdx.x & 31;
  const int wave = threadIdx.x >> 5;
  const int wm = (wave & 1) * 64;
  const int wn = (wave >> 1) * 32;

  stage_tiles(Ag, Bg, sA0, sB0, 0);
  _Float16 *curA = sA0, *curB = sB0, *nxtA = sA1, *nxtB = sB1;

  for (int k0 = 0; k0 < 768; k0 += 64) {
    wait_async0();          // this wave's async copies into cur are done
    __syncthreads();        // everyone's are; everyone finished reading nxt(prev)
    if (k0 + 64 < 768) stage_tiles(Ag, Bg, nxtA, nxtB, k0 + 64);
#pragma unroll
    for (int dc = 0; dc < 2; ++dc) {
      v16h af[4], bf[2];
#pragma unroll
      for (int i = 0; i < 4; ++i)
        af[i] = lds_frag(curA + dc * 32, wm + i * 16 + (lane & 15), LDK);
#pragma unroll
      for (int j = 0; j < 2; ++j)
        bf[j] = lds_frag(curB + dc * 32, wn + j * 16 + (lane & 15), LDK);
#pragma unroll
      for (int i = 0; i < 4; ++i)
#pragma unroll
        for (int j = 0; j < 2; ++j)
          acc[i][j] = WMMA_F16(af[i], bf[j], acc[i][j]);
    }
    _Float16* t;
    t = curA; curA = nxtA; nxtA = t;
    t = curB; curB = nxtB; nxtB = t;
  }
}

// ---------------------------------------------------------------------------
// Fused QKV projection: [4096 x 2304] = Xh[4096x768] @ Wqkv^T, scatter to
// head-major Q/K/V (f16), Q pre-scaled by 1/sqrt(64).
// Section (Q/K/V) is uniform per block: 768 % 128 == 0.
// ---------------------------------------------------------------------------
__global__ __launch_bounds__(256) void qkv_gemm_kernel(
    const _Float16* __restrict__ Xh, const _Float16* __restrict__ Wh,
    const float* __restrict__ bq, const float* __restrict__ bk,
    const float* __restrict__ bv,
    _Float16* __restrict__ Qh, _Float16* __restrict__ Kh, _Float16* __restrict__ Vh) {
  __shared__ __align__(16) _Float16 sA[2][128 * LDK];
  __shared__ __align__(16) _Float16 sB[2][128 * LDK];
  const int m0 = blockIdx.y * 128;
  const int n0 = blockIdx.x * 128;
  v8f acc[4][2] = {};
  gemm128x128(Xh + (size_t)m0 * 768, Wh + (size_t)n0 * 768,
              sA[0], sB[0], sA[1], sB[1], acc);

  // Uniform per-block section decode (no per-element division).
  const int sec = blockIdx.x / 6;                 // 0=Q 1=K 2=V
  const int f0  = n0 - sec * 768;                 // 0..640
  const float* bp = (sec == 0) ? bq : ((sec == 1) ? bk : bv);
  _Float16* dst   = (sec == 0) ? Qh : ((sec == 1) ? Kh : Vh);
  const float scl = (sec == 0) ? 0.125f : 1.0f;   // 1/sqrt(D_K) folded into Q

  const int lane = threadIdx.x & 31;
  const int wave = threadIdx.x >> 5;
  const int wm = (wave & 1) * 64, wn = (wave >> 1) * 32;
  const int colLane = lane & 15;
  const int rowHi = (lane >= 16) ? 8 : 0;
#pragma unroll
  for (int i = 0; i < 4; ++i)
#pragma unroll
    for (int j = 0; j < 2; ++j)
#pragma unroll
      for (int e = 0; e < 8; ++e) {
        const int m = m0 + wm + i * 16 + rowHi + e;
        const int f = f0 + wn + j * 16 + colLane;
        const int h = f >> 6, d = f & 63;
        const int bb = m >> 11, s = m & 2047;
        const float v = (acc[i][j][e] + bp[f]) * scl;
        dst[(((size_t)(bb * 12 + h)) * 2048 + s) * 64 + d] = (_Float16)v;
      }
}

// ---------------------------------------------------------------------------
// Flash attention with the (q%256 >= k%256) mask.
// Grid: x = qmod tile (0..15), y = b*12+h. Block = 256 = 8 waves.
// Wave w handles q rows [w*256 + qb*16, +16): identical mask for all waves,
// so K/V chunk staging is shared block-wide. K/V chunks double-buffered;
// K staged via async-to-LDS, V transposed through lanes.
// ---------------------------------------------------------------------------
__global__ __launch_bounds__(256) void attn_kernel(
    const _Float16* __restrict__ Qh, const _Float16* __restrict__ Kh,
    const _Float16* __restrict__ Vh, _Float16* __restrict__ AttnH) {
  __shared__ __align__(16) _Float16 sK[2][32 * LDK];   // 32 keys x 64 d, row-major
  __shared__ __align__(16) _Float16 sV[2][64 * LDP];   // 64 d x 32 keys, transposed
  __shared__ __align__(16) _Float16 sP[8][16 * LDP];   // per-wave 16x32 probs

  const int tid = threadIdx.x, lane = tid & 31, wave = tid >> 5;
  const int qb = blockIdx.x;                  // 0..15 (qmod tile)
  const int h = blockIdx.y % 12, bb = blockIdx.y / 12;
  const size_t headOff = ((size_t)(bb * 12 + h)) * 2048 * 64;
  const int q0 = wave * 256 + qb * 16;        // this wave's 16 q rows
  const int colLane = lane & 15;
  const int rowHi = (lane >= 16) ? 8 : 0;

  // Resident Q fragments (16 rows x 64 d -> two K=32 chunks), A-layout.
  const _Float16* qrow = Qh + headOff + (size_t)(q0 + colLane) * 64;
  v16h qa[2];
#pragma unroll
  for (int c = 0; c < 2; ++c) qa[c] = glb_frag(qrow + c * 32);

  float mrow[8], lrow[8];
  v8f accO[4] = {};
#pragma unroll
  for (int j = 0; j < 8; ++j) { mrow[j] = NEGINF; lrow[j] = 0.f; }

  const int nk16 = qb + 1;                    // allowed 16-key tiles per window
  const int nch = (nk16 + 1) >> 1;            // 32-key chunks per window
  _Float16* Pt = sP[wave];

  const int sr = tid >> 3;                    // staging: key row 0..31
  const int sc = (tid & 7) * 8;               // staging: d col 0,8,..,56

  // Stage one 32-key chunk (K async row-major, V via lanes transposed).
  auto kick = [&](int wk, int c, int b) {
    const int keyBase = wk * 256 + c * 32;
    const _Float16* kp = Kh + headOff + (size_t)(keyBase + sr) * 64 + sc;
    const _Float16* vp = Vh + headOff + (size_t)(keyBase + sr) * 64 + sc;
    async_b128(&sK[b][sr * LDK + sc], kp);
    v8h vv = *(const v8h*)vp;
#pragma unroll
    for (int e = 0; e < 8; ++e) sV[b][(sc + e) * LDP + sr] = vv[e];
    __builtin_prefetch(vp + 32 * 64, 0, 1);   // next chunk -> global_prefetch
  };

  int kwk = 0, kc = 0;                        // next chunk to kick
  kick(kwk, kc, 0);
  if (++kc == nch) { kc = 0; ++kwk; }
  int buf = 0;

  for (int wk = 0; wk < 8; ++wk) {
    for (int c = 0; c < nch; ++c) {
      wait_async0();
      __syncthreads();          // cur chunk fully staged; prev reads retired
      if (kwk < 8) {
        kick(kwk, kc, buf ^ 1);
        if (++kc == nch) { kc = 0; ++kwk; }
      }
      const _Float16* cK = sK[buf];
      const _Float16* cV = sV[buf];

      const int kt0 = 2 * c, kt1 = 2 * c + 1;
      const bool have1 = (kt1 < nk16);
      const bool diag0 = (kt0 == qb);
      const bool diag1 = have1 && (kt1 == qb);

      // Scores: S = Q (16x64) . K^T (64x16) per key subtile.
      v8f s0 = {}, s1 = {};
#pragma unroll
      for (int dc = 0; dc < 2; ++dc) {
        v16h b0 = lds_frag(cK + dc * 32, colLane, LDK);
        s0 = WMMA_F16(qa[dc], b0, s0);
        if (have1) {
          v16h b1 = lds_frag(cK + dc * 32, 16 + colLane, LDK);
          s1 = WMMA_F16(qa[dc], b1, s1);
        }
      }

      // Online softmax (row stats via shfl reductions in 16-lane halves).
#pragma unroll
      for (int j = 0; j < 8; ++j) {
        float a = s0[j];
        if (diag0 && (j + rowHi) < colLane) a = NEGINF;
        float b = NEGINF;
        if (have1) { b = s1[j]; if (diag1 && (j + rowHi) < colLane) b = NEGINF; }
        float mx = fmaxf(a, b);
#pragma unroll
        for (int msk = 1; msk < 16; msk <<= 1) mx = fmaxf(mx, __shfl_xor(mx, msk, 32));
        const float mnew = fmaxf(mrow[j], mx);
        const float scale = __expf(mrow[j] - mnew);
        mrow[j] = mnew;
        const float e0 = (a <= 0.5f * NEGINF) ? 0.f : __expf(a - mnew);
        const float e1 = (b <= 0.5f * NEGINF) ? 0.f : __expf(b - mnew);
        float rs = e0 + e1;
#pragma unroll
        for (int msk = 1; msk < 16; msk <<= 1) rs += __shfl_xor(rs, msk, 32);
        lrow[j] = lrow[j] * scale + rs;
#pragma unroll
        for (int nt = 0; nt < 4; ++nt) accO[nt][j] *= scale;
        Pt[(j + rowHi) * LDP + colLane]      = (_Float16)e0;
        Pt[(j + rowHi) * LDP + 16 + colLane] = (_Float16)e1;
      }
      asm volatile("s_wait_dscnt 0x0" ::: "memory");   // P visible wave-wide

      // O += P (16x32) . V (32x64)
      v16h pa = lds_frag(Pt, colLane, LDP);
#pragma unroll
      for (int nt = 0; nt < 4; ++nt) {
        v16h vb = lds_frag(cV, nt * 16 + colLane, LDP);
        accO[nt] = WMMA_F16(pa, vb, accO[nt]);
      }
      buf ^= 1;
    }
  }

  // Normalize and store f16 attention output, model-major [B,S,768].
  float inv[8];
#pragma unroll
  for (int j = 0; j < 8; ++j) inv[j] = 1.0f / lrow[j];
  const size_t outBase = ((size_t)bb * 2048) * 768 + (size_t)h * 64;
#pragma unroll
  for (int nt = 0; nt < 4; ++nt)
#pragma unroll
    for (int j = 0; j < 8; ++j) {
      const int r = q0 + rowHi + j;
      const int d = nt * 16 + colLane;
      AttnH[outBase + (size_t)r * 768 + d] = (_Float16)(accO[nt][j] * inv[j]);
    }
}

// ---------------------------------------------------------------------------
// Output projection: out(f32)[4096x768] = AttnH @ Wo^T + bo
// ---------------------------------------------------------------------------
__global__ __launch_bounds__(256) void out_gemm_kernel(
    const _Float16* __restrict__ Ah, const _Float16* __restrict__ Woh,
    const float* __restrict__ bo, float* __restrict__ out) {
  __shared__ __align__(16) _Float16 sA[2][128 * LDK];
  __shared__ __align__(16) _Float16 sB[2][128 * LDK];
  const int m0 = blockIdx.y * 128;
  const int n0 = blockIdx.x * 128;
  v8f acc[4][2] = {};
  gemm128x128(Ah + (size_t)m0 * 768, Woh + (size_t)n0 * 768,
              sA[0], sB[0], sA[1], sB[1], acc);

  const int lane = threadIdx.x & 31;
  const int wave = threadIdx.x >> 5;
  const int wm = (wave & 1) * 64, wn = (wave >> 1) * 32;
  const int colLane = lane & 15;
  const int rowHi = (lane >= 16) ? 8 : 0;
#pragma unroll
  for (int i = 0; i < 4; ++i)
#pragma unroll
    for (int j = 0; j < 2; ++j)
#pragma unroll
      for (int e = 0; e < 8; ++e) {
        const int m = m0 + wm + i * 16 + rowHi + e;
        const int n = n0 + wn + j * 16 + colLane;
        out[(size_t)m * 768 + n] = acc[i][j][e] + bo[n];
      }
}

// ---------------------------------------------------------------------------
// Host launcher
// ---------------------------------------------------------------------------
extern "C" void kernel_launch(void* const* d_in, const int* in_sizes, int n_in,
                              void* d_out, int out_size, void* d_ws, size_t ws_size,
                              hipStream_t stream) {
  const float* x  = (const float*)d_in[0];
  const float* Wq = (const float*)d_in[1];
  const float* bq = (const float*)d_in[2];
  const float* Wk = (const float*)d_in[3];
  const float* bk = (const float*)d_in[4];
  const float* Wv = (const float*)d_in[5];
  const float* bv = (const float*)d_in[6];
  const float* Wo = (const float*)d_in[7];
  const float* bo = (const float*)d_in[8];
  float* out = (float*)d_out;

  // Workspace layout (f16), all 16B aligned.
  _Float16* Xh  = (_Float16*)d_ws;               // 4096*768
  _Float16* Wh  = Xh  + 4096 * 768;              // 2304*768 (Wq|Wk|Wv rows)
  _Float16* Woh = Wh  + 2304 * 768;              // 768*768
  _Float16* Qh  = Woh + 768 * 768;               // 2*12*2048*64
  _Float16* Kh  = Qh  + 3145728;
  _Float16* Vh  = Kh  + 3145728;
  _Float16* Ah  = Vh  + 3145728;                 // 4096*768

  const int NW = 768 * 768;                      // 589824
  cvt_f32_f16<<<(4096 * 768) / 1024, 256, 0, stream>>>(x, Xh, 4096 * 768);
  cvt_f32_f16<<<NW / 1024, 256, 0, stream>>>(Wq, Wh, NW);
  cvt_f32_f16<<<NW / 1024, 256, 0, stream>>>(Wk, Wh + NW, NW);
  cvt_f32_f16<<<NW / 1024, 256, 0, stream>>>(Wv, Wh + 2 * NW, NW);
  cvt_f32_f16<<<NW / 1024, 256, 0, stream>>>(Wo, Woh, NW);

  qkv_gemm_kernel<<<dim3(18, 32), 256, 0, stream>>>(Xh, Wh, bq, bk, bv, Qh, Kh, Vh);
  attn_kernel<<<dim3(16, 24), 256, 0, stream>>>(Qh, Kh, Vh, Ah);
  out_gemm_kernel<<<dim3(6, 32), 256, 0, stream>>>(Ah, Woh, bo, out);
}
